// myModel_75170517614870
// MI455X (gfx1250) — matrix operand order repaired and checked by reading, not compiled, not run on hardware
//
#include <hip/hip_runtime.h>

typedef __attribute__((ext_vector_type(16))) _Float16 v16h;
typedef __attribute__((ext_vector_type(8)))  _Float16 v8h;
typedef __attribute__((ext_vector_type(8)))  float    v8f;
typedef __attribute__((ext_vector_type(2)))  float    v2f;

#define EPS_BN 1e-5f

// ---------------- workspace layout (bytes) ----------------
// [0, 159744)          : 156 packed WMMA B fragments (512 f16 each, per-lane layout)
// [159744, 160000)     : fc1 folded bias (64 f32)
// [160000, +18.9MB)    : f16 feature matrix [16384][576]
#define NFRAG        156
#define FR_B3_BASE   12
#define FR_B4_BASE   36
#define FR_FC1_BASE  84
#define FRQ(i)       ((size_t)(i) * 512)
#define WS_BIAS_OFF  159744
#define WS_FEAT_OFF  160000

struct BranchP {
  const float *w1,*b1,*w2,*b2,*w3,*b3,*w4,*b4;
  const float *g1,*be1,*m1,*v1;
  const float *g2,*be2,*m2,*v2;
  const float *g3,*be3,*m3,*v3;
  const float *g4,*be4,*m4,*v4;
};

#if defined(__has_builtin)
#if __has_builtin(__builtin_amdgcn_wmma_f32_16x16x4_f32)
#define HAVE_WMMA_F32X4 1
#endif
#endif

// var + eps is always >= 1e-5 (well inside normal range) -> bare v_rsq_f32 is exact enough
__device__ __forceinline__ float bn_rsqrt(float v) { return __frsqrt_rn(v + EPS_BN); }

__device__ __forceinline__ v16h load_bfrag(const _Float16* frags, int fi, int lane) {
  return *(const v16h*)(frags + FRQ(fi) + lane * 16);
}

// ------------------------------------------------------------------
// Pack B fragments (weights -> per-lane WMMA register layout, f16).
// k-order for convs: k = t*Cin + ci  (so im2col rows are contiguous).
// fc1 fragments get head-BN1 scale folded in; block NFRAG computes the
// folded fc1 bias  b'[n] = fc1_b[n] + sum_k shift1[k]*W[n,k].
// ------------------------------------------------------------------
__global__ __launch_bounds__(32) void k_pack(
    BranchP P0, BranchP P1,
    const float* __restrict__ fc1w, const float* __restrict__ fc1b,
    const float* __restrict__ hg1, const float* __restrict__ hbe1,
    const float* __restrict__ hm1, const float* __restrict__ hv1,
    _Float16* __restrict__ frags, float* __restrict__ biasf)
{
  const int lane = threadIdx.x;
  const int f = blockIdx.x;
  if (f >= NFRAG) {                       // fc1 folded bias
    for (int n = lane; n < 64; n += 32) {
      float acc = fc1b[n];
      for (int k = 0; k < 576; ++k) {
        float s1 = hg1[k] * bn_rsqrt(hv1[k]);
        acc += (hbe1[k] - hm1[k] * s1) * fc1w[n * 576 + k];
      }
      biasf[n] = acc;
    }
    return;
  }
  const int nlo = lane & 15;
  const int kb = (lane & 16) ? 8 : 0;     // B layout: K = kb + e for e<8, 16 + kb + (e-8) for e>=8
  union { v16h v; _Float16 a[16]; } o;
  if (f < FR_FC1_BASE) {
    const float* W; int Cin, lg, KW, K, nt, kk;
    if (f < FR_B3_BASE) {
      int set = f / 6, rem = f % 6; nt = rem / 3; kk = rem % 3;
      W = set ? P1.w2 : P0.w2; Cin = 16; lg = 4; KW = 5; K = 80;
    } else if (f < FR_B4_BASE) {
      int idx = f - FR_B3_BASE; int set = idx / 12, rem = idx % 12; nt = rem / 3; kk = rem % 3;
      W = set ? P1.w3 : P0.w3; Cin = 32; lg = 5; KW = 3; K = 96;
    } else {
      int idx = f - FR_B4_BASE; int set = idx / 24, rem = idx % 24; nt = rem / 6; kk = rem % 6;
      W = set ? P1.w4 : P0.w4; Cin = 64; lg = 6; KW = 3; K = 192;
    }
    int n = nt * 16 + nlo;
#pragma unroll
    for (int e = 0; e < 16; ++e) {
      int k = kk * 32 + kb + e + (e & 8);
      float v = 0.f;
      if (k < K) {
        int ci = k & (Cin - 1), t = k >> lg;
        v = W[(n * Cin + ci) * KW + t];
      }
      o.a[e] = (_Float16)v;
    }
  } else {                                // fc1 with bn1-head scale folded
    int idx = f - FR_FC1_BASE; int nt = idx / 18, kk = idx % 18;
    int n = nt * 16 + nlo;
#pragma unroll
    for (int e = 0; e < 16; ++e) {
      int k = kk * 32 + kb + e + (e & 8);
      float s1 = hg1[k] * bn_rsqrt(hv1[k]);
      o.a[e] = (_Float16)(fc1w[n * 576 + k] * s1);
    }
  }
  *(v16h*)(frags + FRQ(f) + lane * 16) = o.v;
}

// ------------------------------------------------------------------
// Branch kernel: one wave == 16 samples through one branch-row.
// grid (1024, 3): blockIdx.y = row (0 -> br1 params, 1/2 -> br2).
// ------------------------------------------------------------------
__global__ __launch_bounds__(32) void k_branch(
    const float* __restrict__ x, BranchP P0, BranchP P1,
    const _Float16* __restrict__ frags, _Float16* __restrict__ feat)
{
  const int lane = threadIdx.x;
  const int tile = blockIdx.x;
  const int row  = blockIdx.y;
  const BranchP& p = (row == 0) ? P0 : P1;
  const int set = (row == 0) ? 0 : 1;
  const int nlo = lane & 15;
  const int hf  = lane >> 4;

  alignas(16) __shared__ unsigned char smem[29696];
  float*    sIn   = (float*)smem;              // 16 x 25 f32  (dead after conv1)
  _Float16* sAct2 = (_Float16*)smem;           // 16 x 13 x 32 f16 (aliases sIn)
  _Float16* sAct1 = (_Float16*)(smem + 13312); // 16 x 25 x 16 f16
  _Float16* sAct3 = (_Float16*)(smem + 13312); // 16 x  8 x 64 f16 (aliases sAct1)

  // ---- load one input row (zero-padded by 2 each side) ----
  const float* xb = x + (size_t)tile * 16 * 63 + row * 21;
  for (int s = 0; s < 16; ++s) {
    if (lane < 25) {
      float v = 0.f;
      if (lane >= 2 && lane < 23) v = xb[(size_t)s * 63 + (lane - 2)];
      sIn[s * 25 + lane] = v;
    }
  }
  __syncthreads();

  // ---- conv1 (1->16,kw5,pad2) + bias + relu + bn1 -> sAct1 ----
  {
    float sc = p.g1[nlo] * bn_rsqrt(p.v1[nlo]);
    float sh = p.be1[nlo] - p.m1[nlo] * sc;
    float bias = p.b1[nlo];
    const float hfm = hf ? 0.f : 1.f;      // branch-free masking (all loads unconditional)
#ifdef HAVE_WMMA_F32X4
    v2f b0, b1v;                           // B 4x16 f32: lane<16 K0/K1, lane>=16 K2/K3
    b0[0] = p.w1[nlo * 5 + 2 * hf];
    b0[1] = p.w1[nlo * 5 + 2 * hf + 1];
    b1v[0] = hfm * p.w1[nlo * 5 + 4];
    b1v[1] = 0.f;
    const float* inr = sIn + nlo * 25;
    for (int xo = 0; xo < 21; ++xo) {
      v2f a0, a1;                          // A 16x4 f32 (second wmma covers K=4, rest 0)
      a0[0] = inr[xo + 2 * hf];
      a0[1] = inr[xo + 2 * hf + 1];
      a1[0] = hfm * inr[xo + 4];           // always a valid, finite read; masked by 0/1
      a1[1] = 0.f;
      v8f c = {};
      c = __builtin_amdgcn_wmma_f32_16x16x4_f32(false, a0, false, b0,  (short)0, c, false, false);
      c = __builtin_amdgcn_wmma_f32_16x16x4_f32(false, a1, false, b1v, (short)0, c, false, false);
#pragma unroll
      for (int r = 0; r < 8; ++r) {
        float v = fmaxf(c[r] + bias, 0.f) * sc + sh;
        sAct1[(r + 8 * hf) * 400 + (xo + 2) * 16 + nlo] = (_Float16)v;
      }
    }
#else
    float wv[5];
#pragma unroll
    for (int t = 0; t < 5; ++t) wv[t] = p.w1[nlo * 5 + t];
    for (int r = 0; r < 8; ++r) {
      int sm = 8 * hf + r;
      for (int xo = 0; xo < 21; ++xo) {
        float a = bias;
#pragma unroll
        for (int t = 0; t < 5; ++t) a += wv[t] * sIn[sm * 25 + xo + t];
        float v = fmaxf(a, 0.f) * sc + sh;
        sAct1[sm * 400 + (xo + 2) * 16 + nlo] = (_Float16)v;
      }
    }
#endif
    for (int i = lane; i < 16 * 4 * 16; i += 32) {     // zero pad cols 0,1,23,24
      int sm = i >> 6, rem = i & 63, cs = rem >> 4, ci = rem & 15;
      int col = (cs == 0) ? 0 : (cs == 1) ? 1 : (cs == 2) ? 23 : 24;
      sAct1[sm * 400 + col * 16 + ci] = (_Float16)0.f;
    }
  }
  __syncthreads();

  // ---- conv2 (16->32,kw5,pad2) + bn2 -> maxpool(w2,pad1) -> relu -> sAct2 ----
  {
    v16h B[2][3];
#pragma unroll
    for (int nt = 0; nt < 2; ++nt)
#pragma unroll
      for (int kk = 0; kk < 3; ++kk)
        B[nt][kk] = load_bfrag(frags, (set * 2 + nt) * 3 + kk, lane);
    float sc[2], sh[2];
#pragma unroll
    for (int nt = 0; nt < 2; ++nt) {
      int n = nt * 16 + nlo;
      float s1 = p.g2[n] * bn_rsqrt(p.v2[n]);
      sc[nt] = s1;
      sh[nt] = p.be2[n] - p.m2[n] * s1 + p.b2[n] * s1;
    }
    const _Float16* ab = sAct1 + nlo * 400;
    const int klo8 = (lane & 16) ? 8 : 0;
    const v8h Z8 = {};
    v8f pc0 = {}, pc1 = {};
    for (int xo = 0; xo < 21; ++xo) {
      v8f acc0 = {}, acc1 = {};
#pragma unroll
      for (int kk = 0; kk < 3; ++kk) {                  // K=80 padded to 96
        union { v16h v; v8h h[2]; } u;
        int klo = (kk << 5) + klo8;
        u.h[0] = *(const v8h*)(ab + xo * 16 + klo);
        u.h[1] = (kk == 2) ? Z8 : *(const v8h*)(ab + xo * 16 + klo + 16);
        acc0 = __builtin_amdgcn_wmma_f32_16x16x32_f16(false, u.v, false, B[0][kk], (short)0, acc0, false, false);
        acc1 = __builtin_amdgcn_wmma_f32_16x16x32_f16(false, u.v, false, B[1][kk], (short)0, acc1, false, false);
      }
      v8f w0, w1v;
#pragma unroll
      for (int r = 0; r < 8; ++r) { w0[r] = acc0[r] * sc[0] + sh[0]; w1v[r] = acc1[r] * sc[1] + sh[1]; }
      if (xo & 1) { pc0 = w0; pc1 = w1v; }              // pooled j covers {2j-1, 2j}
      else {
        int j = xo >> 1;
#pragma unroll
        for (int r = 0; r < 8; ++r) {
          float m0 = (xo == 0) ? w0[r]  : fmaxf(pc0[r], w0[r]);
          float m1 = (xo == 0) ? w1v[r] : fmaxf(pc1[r], w1v[r]);
          int sm = r + 8 * hf;
          sAct2[sm * 416 + (j + 1) * 32 + nlo]      = (_Float16)fmaxf(m0, 0.f);
          sAct2[sm * 416 + (j + 1) * 32 + 16 + nlo] = (_Float16)fmaxf(m1, 0.f);
        }
      }
    }
    for (int i = lane; i < 16 * 2 * 32; i += 32) {      // zero pad cols 0,12
      int sm = i >> 6, rem = i & 63, cs = rem >> 5, ci = rem & 31;
      sAct2[sm * 416 + (cs ? 12 : 0) * 32 + ci] = (_Float16)0.f;
    }
  }
  __syncthreads();

  // ---- conv3 (32->64,kw3,pad1) + bn3 -> maxpool(w2,pad1) -> relu -> sAct3 ----
  {
    v16h B[4][3];
#pragma unroll
    for (int nt = 0; nt < 4; ++nt)
#pragma unroll
      for (int kk = 0; kk < 3; ++kk)
        B[nt][kk] = load_bfrag(frags, FR_B3_BASE + (set * 4 + nt) * 3 + kk, lane);
    float sc[4], sh[4];
#pragma unroll
    for (int nt = 0; nt < 4; ++nt) {
      int n = nt * 16 + nlo;
      float s1 = p.g3[n] * bn_rsqrt(p.v3[n]);
      sc[nt] = s1;
      sh[nt] = p.be3[n] - p.m3[n] * s1 + p.b3[n] * s1;
    }
    const _Float16* ab = sAct2 + nlo * 416;
    const int klo8 = (lane & 16) ? 8 : 0;
    v8f pc[4];
    v8f z0 = {};
#pragma unroll
    for (int nt = 0; nt < 4; ++nt) pc[nt] = z0;
    for (int xo = 0; xo < 11; ++xo) {
      v8f acc[4];
#pragma unroll
      for (int nt = 0; nt < 4; ++nt) acc[nt] = z0;
#pragma unroll
      for (int kk = 0; kk < 3; ++kk) {                  // K=96 exact
        union { v16h v; v8h h[2]; } u;
        int klo = (kk << 5) + klo8;
        u.h[0] = *(const v8h*)(ab + xo * 32 + klo);
        u.h[1] = *(const v8h*)(ab + xo * 32 + klo + 16);
#pragma unroll
        for (int nt = 0; nt < 4; ++nt)
          acc[nt] = __builtin_amdgcn_wmma_f32_16x16x32_f16(false, u.v, false, B[nt][kk], (short)0, acc[nt], false, false);
      }
#pragma unroll
      for (int nt = 0; nt < 4; ++nt) {
        v8f w;
#pragma unroll
        for (int r = 0; r < 8; ++r) w[r] = acc[nt][r] * sc[nt] + sh[nt];
        if (xo & 1) pc[nt] = w;
        else {
          int j = xo >> 1;
#pragma unroll
          for (int r = 0; r < 8; ++r) {
            float m = (xo == 0) ? w[r] : fmaxf(pc[nt][r], w[r]);
            sAct3[(r + 8 * hf) * 512 + (j + 1) * 64 + nt * 16 + nlo] = (_Float16)fmaxf(m, 0.f);
          }
        }
      }
    }
    for (int i = lane; i < 16 * 2 * 64; i += 32) {      // zero pad cols 0,7
      int sm = i >> 7, rem = i & 127, cs = rem >> 6, ci = rem & 63;
      sAct3[sm * 512 + (cs ? 7 : 0) * 64 + ci] = (_Float16)0.f;
    }
  }
  __syncthreads();

  // ---- conv4 (64->64,kw3,pad1) + bn4 -> avgpool(w2) -> relu -> feat (global, f16) ----
  {
    const _Float16* ab = sAct3 + nlo * 512;
    const int klo8 = (lane & 16) ? 8 : 0;
    _Float16* fb = feat + ((size_t)tile * 16) * 576 + row * 192;
    v8f z0 = {};
    for (int nt = 0; nt < 4; ++nt) {
      v8f acc[6];
#pragma unroll
      for (int j = 0; j < 6; ++j) acc[j] = z0;
      for (int kk = 0; kk < 6; ++kk) {                  // K=192 exact
        v16h Bf = load_bfrag(frags, FR_B4_BASE + (set * 4 + nt) * 6 + kk, lane);
        int klo = (kk << 5) + klo8;
#pragma unroll
        for (int xo = 0; xo < 6; ++xo) {
          union { v16h v; v8h h[2]; } u;
          u.h[0] = *(const v8h*)(ab + xo * 64 + klo);
          u.h[1] = *(const v8h*)(ab + xo * 64 + klo + 16);
          acc[xo] = __builtin_amdgcn_wmma_f32_16x16x32_f16(false, u.v, false, Bf, (short)0, acc[xo], false, false);
        }
      }
      int n = nt * 16 + nlo;
      float s1 = p.g4[n] * bn_rsqrt(p.v4[n]);
      float sh = p.be4[n] - p.m4[n] * s1 + p.b4[n] * s1;
#pragma unroll
      for (int j = 0; j < 3; ++j) {
#pragma unroll
        for (int r = 0; r < 8; ++r) {
          float a = acc[2 * j][r] * s1 + sh;
          float b = acc[2 * j + 1][r] * s1 + sh;
          float v = fmaxf(0.5f * (a + b), 0.f);
          fb[(size_t)(r + 8 * hf) * 576 + n * 3 + j] = (_Float16)v;  // feature = ch*3 + j
        }
      }
    }
  }
}

// ------------------------------------------------------------------
// Head: (bn1 folded into packed fc1 B/bias) -> fc1 WMMA -> bn2 -> fc2.
// ------------------------------------------------------------------
__global__ __launch_bounds__(32) void k_head(
    const _Float16* __restrict__ feat, const _Float16* __restrict__ frags,
    const float* __restrict__ biasf,
    const float* __restrict__ hg2, const float* __restrict__ hbe2,
    const float* __restrict__ hm2, const float* __restrict__ hv2,
    const float* __restrict__ fc2w, const float* __restrict__ fc2b,
    float* __restrict__ out)
{
  const int lane = threadIdx.x;
  const int tile = blockIdx.x;
  const int nlo = lane & 15, hf = lane >> 4;
  __shared__ float h2[16][64];
  const _Float16* ab = feat + ((size_t)tile * 16 + nlo) * 576;
  const int klo8 = (lane & 16) ? 8 : 0;
  for (int nt = 0; nt < 4; ++nt) {
    v8f acc = {};
    for (int kk = 0; kk < 18; ++kk) {                   // K=576 exact
      v16h Bf = load_bfrag(frags, FR_FC1_BASE + nt * 18 + kk, lane);
      union { v16h v; v8h h[2]; } u;
      int klo = (kk << 5) + klo8;
      u.h[0] = *(const v8h*)(ab + klo);
      u.h[1] = *(const v8h*)(ab + klo + 16);
      acc = __builtin_amdgcn_wmma_f32_16x16x32_f16(false, u.v, false, Bf, (short)0, acc, false, false);
    }
    int n = nt * 16 + nlo;
    float s2 = hg2[n] * bn_rsqrt(hv2[n]);
    float s0 = hbe2[n] - hm2[n] * s2;
    float bias = biasf[n];
#pragma unroll
    for (int r = 0; r < 8; ++r) h2[r + 8 * hf][n] = (acc[r] + bias) * s2 + s0;
  }
  __syncthreads();
  for (int t = lane; t < 64; t += 32) {                 // fc2: 16 samples x 4 outs
    int s = t >> 2, o = t & 3;
    float a = fc2b[o];
#pragma unroll 8
    for (int n = 0; n < 64; ++n) a += h2[s][n] * fc2w[o * 64 + n];
    out[((size_t)tile * 16 + s) * 4 + o] = a;
  }
}

extern "C" void kernel_launch(void* const* d_in, const int* in_sizes, int n_in,
                              void* d_out, int out_size, void* d_ws, size_t ws_size,
                              hipStream_t stream) {
  (void)in_sizes; (void)n_in; (void)out_size; (void)ws_size;
  auto F = [&](int i) { return (const float*)d_in[i]; };
  const float* x = F(0);
  BranchP b1 = { F(1),F(2),F(3),F(4),F(5),F(6),F(7),F(8),
                 F(9),F(10),F(11),F(12), F(13),F(14),F(15),F(16),
                 F(17),F(18),F(19),F(20), F(21),F(22),F(23),F(24) };
  BranchP b2 = { F(25),F(26),F(27),F(28),F(29),F(30),F(31),F(32),
                 F(33),F(34),F(35),F(36), F(37),F(38),F(39),F(40),
                 F(41),F(42),F(43),F(44), F(45),F(46),F(47),F(48) };
  const float* fc1w = F(49); const float* fc1b = F(50);
  const float* fc2w = F(51); const float* fc2b = F(52);
  const float* hg1 = F(53); const float* hbe1 = F(54); const float* hm1 = F(55); const float* hv1 = F(56);
  const float* hg2 = F(57); const float* hbe2 = F(58); const float* hm2 = F(59); const float* hv2 = F(60);

  _Float16* frags = (_Float16*)d_ws;
  float*    biasf = (float*)((char*)d_ws + WS_BIAS_OFF);
  _Float16* feat  = (_Float16*)((char*)d_ws + WS_FEAT_OFF);

  k_pack<<<NFRAG + 1, 32, 0, stream>>>(b1, b2, fc1w, fc1b, hg1, hbe1, hm1, hv1, frags, biasf);
  k_branch<<<dim3(1024, 3, 1), 32, 0, stream>>>(x, b1, b2, frags, feat);
  k_head<<<1024, 32, 0, stream>>>(feat, frags, biasf, hg2, hbe2, hm2, hv2, fc2w, fc2b, (float*)d_out);
}
